// PostProcess_3934190044268
// MI455X (gfx1250) — 8-bit matrix operand order repaired and checked by reading, compiled, not run on hardware
//
#include <hip/hip_runtime.h>
#include <stdint.h>

#define B_IMG   32
#define N_PRI   8732
#define C_CLS   81
#define CM1     80
#define K_NMS   1000
#define DET     100
#define SZ      16384
#define NW      32
#define SCORE_THRESH 0.5f
#define NMS_THRESH   0.45f
#define MIN_SIZE     0.01f
#define V0f 0.1f
#define V1f 0.2f

typedef int v8i  __attribute__((ext_vector_type(8)));
typedef int vsi4 __attribute__((vector_size(16)));

#define AS1 __attribute__((address_space(1)))
#define AS3 __attribute__((address_space(3)))

#if defined(__AMDGCN__) && \
    __has_builtin(__builtin_amdgcn_global_load_async_to_lds_b128) && \
    __has_builtin(__builtin_amdgcn_global_load_async_to_lds_b32)  && \
    __has_builtin(__builtin_amdgcn_s_wait_asynccnt)
#define USE_ASYNC_LDS 1
#else
#define USE_ASYNC_LDS 0
#endif

__global__ void k_decode_softmax(const float* __restrict__ logits,
                                 const float* __restrict__ loc,
                                 const float* __restrict__ priors,
                                 float* __restrict__ dec_boxes,
                                 unsigned long long* __restrict__ keys) {
    int t = blockIdx.x * blockDim.x + threadIdx.x;
    if (t >= B_IMG * SZ) return;
    int b = t / SZ, n = t % SZ;
    unsigned long long key = ~0ull;
    if (n < N_PRI) {
        const float* lg = logits + ((size_t)b * N_PRI + n) * C_CLS;
        __builtin_prefetch(lg, 0, 1);
        float m = lg[0];
        float bestv = -1e30f; int bestc = 1;
        #pragma unroll 4
        for (int c = 1; c < C_CLS; ++c) {
            float v = lg[c];
            m = fmaxf(m, v);
            if (v > bestv) { bestv = v; bestc = c; }
        }
        float s = 0.f;
        #pragma unroll 4
        for (int c = 0; c < C_CLS; ++c) s += expf(lg[c] - m);
        float score = expf(bestv - m) / s;

        const float* lc = loc + ((size_t)b * N_PRI + n) * 4;
        float px = priors[n*4+0], py = priors[n*4+1];
        float pw = priors[n*4+2], ph = priors[n*4+3];
        float cx = px + lc[0] * V0f * pw;
        float cy = py + lc[1] * V0f * ph;
        float w  = pw * expf(lc[2] * V1f);
        float h  = ph * expf(lc[3] * V1f);
        float x1 = cx - 0.5f*w, y1 = cy - 0.5f*h;
        float x2 = cx + 0.5f*w, y2 = cy + 0.5f*h;
        float* db = dec_boxes + ((size_t)b * N_PRI + n) * 4;
        db[0] = x1; db[1] = y1; db[2] = x2; db[3] = y2;

        bool size_ok = (x2 - x1 >= MIN_SIZE) && (y2 - y1 >= MIN_SIZE);
        if (score > SCORE_THRESH && size_ok) {
            unsigned u  = __float_as_uint(score);
            unsigned mm = u ^ ((u >> 31) ? 0xFFFFFFFFu : 0x80000000u);
            unsigned skey = ~mm;
            unsigned idx  = (unsigned)(n * CM1 + (bestc - 1));
            key = ((unsigned long long)skey << 32) | idx;
        }
    }
    keys[(size_t)b * SZ + n] = key;
}

__global__ void k_bitonic(unsigned long long* __restrict__ keys, int j, int k) {
    int t = blockIdx.x * blockDim.x + threadIdx.x;
    if (t >= B_IMG * SZ) return;
    int seg = t / SZ, i = t % SZ;
    int p = i ^ j;
    if (p > i) {
        unsigned long long* base = keys + (size_t)seg * SZ;
        unsigned long long a = base[i], c = base[p];
        bool asc = ((i & k) == 0);
        bool doswap = asc ? (a > c) : (a < c);
        if (doswap) { base[i] = c; base[p] = a; }
    }
}

__global__ void k_extract(const unsigned long long* __restrict__ keys,
                          const float* __restrict__ dec_boxes,
                          float* __restrict__ cs, int* __restrict__ cl,
                          float* __restrict__ cb) {
    int t = blockIdx.x * blockDim.x + threadIdx.x;
    if (t >= B_IMG * K_NMS) return;
    int b = t / K_NMS;
    unsigned long long kk = keys[(size_t)b * SZ + (t % K_NMS)];
    unsigned skey = (unsigned)(kk >> 32);
    unsigned idx  = (unsigned)kk;
    bool valid = (skey != 0xFFFFFFFFu);
    float score = -1.f; int lbl = 0;
    float bx0 = 0.f, bx1 = 0.f, bx2 = 0.f, bx3 = 0.f;
    if (valid) {
        unsigned m = ~skey;
        unsigned u = (m & 0x80000000u) ? (m ^ 0x80000000u) : ~m;
        score = __uint_as_float(u);
        lbl   = (int)(idx % CM1) + 1;
        const float* db = dec_boxes + ((size_t)b * N_PRI + idx / CM1) * 4;
        bx0 = db[0]; bx1 = db[1]; bx2 = db[2]; bx3 = db[3];
    }
    cs[t] = score; cl[t] = lbl;
    float* o = cb + (size_t)t * 4;
    o[0] = bx0; o[1] = bx1; o[2] = bx2; o[3] = bx3;
}

__global__ void k_zero(unsigned* __restrict__ p, int n) {
    int t = blockIdx.x * blockDim.x + threadIdx.x;
    if (t < n) p[t] = 0u;
}

__global__ __launch_bounds__(32)
void k_supmat(const float* __restrict__ cb, const int* __restrict__ cl,
              unsigned* __restrict__ maskw) {
    const int tc = blockIdx.x, tr = blockIdx.y, b = blockIdx.z;
    if (tc < tr) return;

    __shared__ float boxs[32][4];
    __shared__ int   lbls[32];
    const int lane = threadIdx.x;
    const int half = lane >> 4;
    const int m16  = lane & 15;

    const int tile = half ? tc : tr;
    const int ci   = tile * 16 + m16;
    const bool ok  = (ci < K_NMS);
    const int cidx = ok ? ci : (K_NMS - 1);

#if USE_ASYNC_LDS
    {
        const float* bsrc = cb + ((size_t)b * K_NMS + cidx) * 4;
        const int*   lsrc = cl + b * K_NMS + cidx;
        __builtin_amdgcn_global_load_async_to_lds_b128(
            (AS1 vsi4*)(void*)bsrc, (AS3 vsi4*)(void*)&boxs[lane][0], 0, 0);
        __builtin_amdgcn_global_load_async_to_lds_b32(
            (AS1 int*)(void*)lsrc, (AS3 int*)(void*)&lbls[lane], 0, 0);
        __builtin_amdgcn_s_wait_asynccnt(0);
        if (!ok) {
            boxs[lane][0] = 0.f; boxs[lane][1] = 0.f;
            boxs[lane][2] = 0.f; boxs[lane][3] = 0.f;
            lbls[lane] = 0;
        }
    }
#else
    {
        lbls[lane] = ok ? cl[b * K_NMS + cidx] : 0;
        for (int q = 0; q < 4; ++q)
            boxs[lane][q] = ok ? cb[((size_t)b * K_NMS + cidx) * 4 + q] : 0.f;
    }
#endif
    __syncthreads();

    const int lblA = lbls[m16];
    const int lblB = lbls[16 + m16];

    v8i acc = {};
    #pragma unroll
    for (int koff = 0; koff < 128; koff += 64) {
        v8i A, Bm;
        const int la = lblA - koff, lb = lblB - koff;
        #pragma unroll
        for (int v = 0; v < 8; ++v) {
            const int abase = ((v >> 2) << 5) + ((v & 3) << 2) + (half << 4);
            const int bbase = ((v >> 2) << 5) + ((v & 3) << 2) + (half << 4);
            const unsigned da = (unsigned)(la - abase);
            const unsigned db = (unsigned)(lb - bbase);
            A[v]  = (da < 4u) ? (1 << (int)(da << 3)) : 0;
            Bm[v] = (db < 4u) ? (1 << (int)(db << 3)) : 0;
        }
        acc = __builtin_amdgcn_wmma_i32_16x16x64_iu8(false, A, false, Bm, acc,
                                                     false, false);
    }

    const int j = tc * 16 + m16;
    if (j < K_NMS) {
        const float cx1 = boxs[16 + m16][0], cy1 = boxs[16 + m16][1];
        const float cx2 = boxs[16 + m16][2], cy2 = boxs[16 + m16][3];
        const float carea = (cx2 - cx1) * (cy2 - cy1);
        #pragma unroll
        for (int d = 0; d < 8; ++d) {
            const int il = d + (half << 3);
            const int i  = tr * 16 + il;
            if (i < K_NMS && j > i && acc[d] != 0) {
                float rx1 = boxs[il][0], ry1 = boxs[il][1];
                float rx2 = boxs[il][2], ry2 = boxs[il][3];
                float ix1 = fmaxf(rx1, cx1), iy1 = fmaxf(ry1, cy1);
                float ix2 = fminf(rx2, cx2), iy2 = fminf(ry2, cy2);
                float inter = fmaxf(ix2 - ix1, 0.f) * fmaxf(iy2 - iy1, 0.f);
                float rarea = (rx2 - rx1) * (ry2 - ry1);
                float uni   = fmaxf(rarea + carea - inter, 1e-9f);
                if (inter / uni > NMS_THRESH) {
                    atomicOr(&maskw[((size_t)b * K_NMS + i) * NW + (j >> 5)],
                             1u << (j & 31));
                }
            }
        }
    }
}

__global__ __launch_bounds__(1024)
void k_nms_out(const float* __restrict__ cs, const int* __restrict__ cl,
               const float* __restrict__ cb, const unsigned* __restrict__ maskw,
               const float* __restrict__ tsz,
               float* __restrict__ boxes_out, float* __restrict__ scores_out,
               int* __restrict__ labels_out) {
    const int b = blockIdx.x;
    const int t = threadIdx.x;
    __shared__ unsigned char keep[K_NMS];
    __shared__ int ord[DET];
    __shared__ int cnt;

    if (t < K_NMS) keep[t] = (cs[b * K_NMS + t] > 0.f) ? 1 : 0;
    __syncthreads();

    for (int i = 0; i < K_NMS; ++i) {
        if (keep[i]) {
            if (t > i && t < K_NMS) {
                unsigned w = maskw[((size_t)b * K_NMS + i) * NW + (t >> 5)];
                if ((w >> (t & 31)) & 1u) keep[t] = 0;
            }
        }
        __syncthreads();
    }

    if (t == 0) {
        int c = 0;
        for (int i = 0; i < K_NMS && c < DET; ++i)
            if (keep[i]) ord[c++] = i;
        cnt = c;
    }
    __syncthreads();

    const float s0 = tsz[b * 2 + 0], s1 = tsz[b * 2 + 1];
    const float sc[4] = { s1, s0, s1, s0 };
    if (t < DET) {
        float bx[4] = {0.f, 0.f, 0.f, 0.f};
        float s = 0.f; int l = 0;
        if (t < cnt) {
            int r = ord[t];
            s = cs[b * K_NMS + r];
            l = cl[b * K_NMS + r];
            #pragma unroll
            for (int q = 0; q < 4; ++q)
                bx[q] = cb[((size_t)b * K_NMS + r) * 4 + q] * sc[q];
        }
        float* bo = boxes_out + ((size_t)b * DET + t) * 4;
        bo[0] = bx[0]; bo[1] = bx[1]; bo[2] = bx[2]; bo[3] = bx[3];
        scores_out[b * DET + t] = s;
        labels_out[b * DET + t] = l;
    }
}

extern "C" void kernel_launch(void* const* d_in, const int* in_sizes, int n_in,
                              void* d_out, int out_size, void* d_ws, size_t ws_size,
                              hipStream_t stream) {
    (void)in_sizes; (void)n_in; (void)out_size; (void)ws_size;
    const float* logits = (const float*)d_in[0];
    const float* loc    = (const float*)d_in[1];
    const float* priors = (const float*)d_in[2];
    const float* tsz    = (const float*)d_in[3];

    char* ws = (char*)d_ws;
    size_t off = 0;
    auto take = [&](size_t bytes) { size_t o = off; off += (bytes + 255) & ~(size_t)255; return o; };
    unsigned long long* keys = (unsigned long long*)(ws + take((size_t)B_IMG * SZ * 8));
    float*    dec  = (float*)(ws + take((size_t)B_IMG * N_PRI * 4 * 4));
    float*    cs   = (float*)(ws + take((size_t)B_IMG * K_NMS * 4));
    int*      clp  = (int*)  (ws + take((size_t)B_IMG * K_NMS * 4));
    float*    cbp  = (float*)(ws + take((size_t)B_IMG * K_NMS * 4 * 4));
    unsigned* mwp  = (unsigned*)(ws + take((size_t)B_IMG * K_NMS * NW * 4));

    float* boxes_out  = (float*)d_out;
    float* scores_out = boxes_out + (size_t)B_IMG * DET * 4;
    int*   labels_out = (int*)(scores_out + (size_t)B_IMG * DET);

    {
        int nthr = B_IMG * SZ;
        k_decode_softmax<<<(nthr + 255) / 256, 256, 0, stream>>>(
            logits, loc, priors, dec, keys);
    }
    for (int k = 2; k <= SZ; k <<= 1)
        for (int j = k >> 1; j > 0; j >>= 1) {
            int nthr = B_IMG * SZ;
            k_bitonic<<<(nthr + 255) / 256, 256, 0, stream>>>(keys, j, k);
        }
    {
        int nthr = B_IMG * K_NMS;
        k_extract<<<(nthr + 255) / 256, 256, 0, stream>>>(keys, dec, cs, clp, cbp);
    }
    {
        int nthr = B_IMG * K_NMS * NW;
        k_zero<<<(nthr + 255) / 256, 256, 0, stream>>>(mwp, nthr);
    }
    {
        dim3 grid(63, 63, B_IMG);
        k_supmat<<<grid, 32, 0, stream>>>(cbp, clp, mwp);
    }
    k_nms_out<<<B_IMG, 1024, 0, stream>>>(cs, clp, cbp, mwp, tsz,
                                          boxes_out, scores_out, labels_out);
}
